// WindowAttention_12128987644017
// MI455X (gfx1250) — matrix operand order repaired and checked by reading, not compiled, not run on hardware
//
#include <hip/hip_runtime.h>
#include <hip/hip_bf16.h>

typedef __attribute__((ext_vector_type(16))) __bf16 v16bf;
typedef __attribute__((ext_vector_type(8)))  __bf16 v8bf;
typedef __attribute__((ext_vector_type(4)))  __bf16 v4bf;
typedef __attribute__((ext_vector_type(8)))  float  v8f;

#define DIM     192
#define NH      6
#define HD      32
#define NTOK    64
#define NWIN    4096
#define NW_IMG  1024
#define QKV_OUT 576

// ---------- LDS layout ----------
#define XSTR 200   // x / o staging row stride (bf16 elems), 400B rows (16B mult)
#define QSTR 40    // q/k row stride (bf16), 80B
#define VSTR 72    // v-transposed row stride (bf16), 144B
#define PSTR 72    // per-wave softmax P row stride (bf16), 144B
#define OFF_XS 0
#define OFF_QS 25600    // 64*200*2
#define OFF_KS 56320    // +6*64*40*2
#define OFF_VT 87040    // +6*64*40*2
#define OFF_PW 114688   // +6*32*72*2
#define SMEM_BYTES 133120  // +8 waves * 16*72*2  -> 2 workgroups / WGP

__device__ __forceinline__ v8f vz8() {
    v8f z = {0.f,0.f,0.f,0.f,0.f,0.f,0.f,0.f};
    return z;
}

// A-fragment (16x32 bf16): lane L holds row (rcbase + L%16);
// elems 0..15 = K kbase+koff+0..15, koff = (L>=16)?16:0
__device__ __forceinline__ v16bf load_fragA(const __bf16* base, int stride,
                                            int rcbase, int lane, int kbase) {
    const __bf16* p = base + (rcbase + (lane & 15)) * stride + kbase + ((lane >> 4) << 4);
    v8bf lo = *(const v8bf*)p;
    v8bf hi = *(const v8bf*)(p + 8);
    return __builtin_shufflevector(lo, hi, 0,1,2,3,4,5,6,7,8,9,10,11,12,13,14,15);
}

// B-fragment (32x16 bf16): lane L holds column (nbase + L%16);
// elems 0..15 = K kbase + ((L>=16)?16:0) + 0..15 (contiguous in row-major [n][k] storage)
__device__ __forceinline__ v16bf load_fragB(const __bf16* base, int stride,
                                            int nbase, int lane, int kbase) {
    const __bf16* p = base + (nbase + (lane & 15)) * stride + kbase + ((lane >> 4) << 4);
    v8bf lo = *(const v8bf*)p;
    v8bf hi = *(const v8bf*)(p + 8);
    return __builtin_shufflevector(lo, hi, 0,1,2,3,4,5,6,7,8,9,10,11,12,13,14,15);
}

// Pre-packed weight block: 32 lanes * 16 bf16, lane-contiguous (32B per lane)
__device__ __forceinline__ v16bf load_wblk(const __bf16* blocks, int blk, int lane) {
    const __bf16* p = blocks + blk * 512 + lane * 16;
    v8bf lo = *(const v8bf*)p;
    v8bf hi = *(const v8bf*)(p + 8);
    return __builtin_shufflevector(lo, hi, 0,1,2,3,4,5,6,7,8,9,10,11,12,13,14,15);
}

// ---------------- prep: pack fp32 [K, ncols] weights into B-fragment bf16 blocks ----------------
__global__ void pack_w_kernel(const float* __restrict__ W, __bf16* __restrict__ out,
                              int ncols, int nt_count) {
    int blk  = blockIdx.x;      // kt * nt_count + nt
    int lane = threadIdx.x;     // 0..31
    int kt = blk / nt_count, nt = blk % nt_count;
    __bf16 vals[16];
    int kofs = kt * 32 + ((lane >> 4) << 4);
    int col  = nt * 16 + (lane & 15);
    #pragma unroll
    for (int e = 0; e < 16; ++e)
        vals[e] = (__bf16)W[(kofs + e) * ncols + col];
    __bf16* dst = out + blk * 512 + lane * 16;
    *(v8bf*)dst       = *(v8bf*)&vals[0];
    *(v8bf*)(dst + 8) = *(v8bf*)&vals[8];
}

// ---------------- prep: expand relative position bias to dense [NH,64,64] f32 ----------------
__global__ void rpb_kernel(const float* __restrict__ rpb_table,
                           const int* __restrict__ rel_idx,
                           float* __restrict__ out) {
    int t = blockIdx.x * blockDim.x + threadIdx.x;   // 6*4096
    if (t >= NH * NTOK * NTOK) return;
    int h = t >> 12;
    int r = t & 4095;
    out[t] = rpb_table[rel_idx[r] * NH + h];
}

// ---------------- main fused window-attention kernel (1 block = 1 window) ----------------
__global__ void __launch_bounds__(256)
win_attn_kernel(const float* __restrict__ x, const float* __restrict__ mask,
                const __bf16* __restrict__ wqkv, const float* __restrict__ bqkv,
                const __bf16* __restrict__ wproj, const float* __restrict__ bproj,
                const float* __restrict__ rpbf, float* __restrict__ out) {
    extern __shared__ char smem[];
    __bf16* xs = (__bf16*)(smem + OFF_XS);  // [64][XSTR] x staging; reused as O staging
    __bf16* qs = (__bf16*)(smem + OFF_QS);  // [NH][64][QSTR] q (pre-scaled)
    __bf16* ks = (__bf16*)(smem + OFF_KS);  // [NH][64][QSTR]
    __bf16* vt = (__bf16*)(smem + OFF_VT);  // [NH][32][VSTR] v transposed
    __bf16* pw = (__bf16*)(smem + OFF_PW);  // [8 waves][16][PSTR] per-wave P scratch

    const int tid  = threadIdx.x;
    const int lane = tid & 31;
    // wave index is wave-uniform: pin it to an SGPR so all tile index math is scalar
    const int wave = __builtin_amdgcn_readfirstlane(tid >> 5);
    const int w    = blockIdx.x;
    const int n_lo = lane & 15;
    const int rhi  = (lane >> 4) << 3;
    const float scale = 0.17677669529663687f;  // 32^-0.5

    // -------- Phase A: stage x -> bf16 LDS (b64 packed stores) --------
    const float* xw = x + (size_t)w * NTOK * DIM;
    #pragma unroll
    for (int i = 0; i < 12; ++i) {
        int j   = tid + 256 * i;           // 3072 float4 chunks
        int row = j / 48;
        int c4  = (j % 48) * 4;
        float4 v = *(const float4*)(xw + row * DIM + c4);
        v4bf pk = { (__bf16)v.x, (__bf16)v.y, (__bf16)v.z, (__bf16)v.w };
        *(v4bf*)(xs + row * XSTR + c4) = pk;
    }
    __syncthreads();

    // -------- Phase B: QKV = x @ Wqkv + b, scatter q/k/vT per head --------
    #pragma unroll 1
    for (int ti = 0; ti < 18; ++ti) {
        int t  = wave + 8 * ti;            // scalar: 144 tiles = 4 mtiles x 36 ntiles
        int mt = t / 36, nt = t % 36;
        v8f acc = vz8();
        #pragma unroll
        for (int kt = 0; kt < 6; ++kt) {
            v16bf A = load_fragA(xs, XSTR, mt * 16, lane, kt * 32);
            v16bf B = load_wblk(wqkv, kt * 36 + nt, lane);
            acc = __builtin_amdgcn_wmma_f32_16x16x32_bf16(false, A, false, B,
                                                          (short)0, acc, false, false);
        }
        // scalar (tile-uniform) destination decode -> uniform branch, no exec juggling
        int kind  = nt / 12;               // 0=q 1=k 2=v
        int nt12  = nt % 12;
        int h     = nt12 >> 1;
        int dbase = (nt12 & 1) << 4;
        int d     = dbase + n_lo;
        float bias = bqkv[nt * 16 + n_lo];
        int rowb  = mt * 16 + rhi;
        if (kind == 0) {
            __bf16* p = qs + (h * 64 + rowb) * QSTR + d;
            #pragma unroll
            for (int r = 0; r < 8; ++r)
                p[r * QSTR] = (__bf16)((acc[r] + bias) * scale);
        } else if (kind == 1) {
            __bf16* p = ks + (h * 64 + rowb) * QSTR + d;
            #pragma unroll
            for (int r = 0; r < 8; ++r)
                p[r * QSTR] = (__bf16)(acc[r] + bias);
        } else {
            __bf16* p = vt + (h * 32 + d) * VSTR + rowb;
            #pragma unroll
            for (int r = 0; r < 8; ++r)
                p[r] = (__bf16)(acc[r] + bias);
        }
    }
    __syncthreads();

    // -------- Phase C+D: attention as 24 (head, mtile) units over all 8 waves --------
    const float* maskw = mask + (size_t)(w & (NW_IMG - 1)) * (NTOK * NTOK);
    __bf16* ph = pw + wave * (16 * PSTR);  // private per-wave P scratch
    #pragma unroll 1
    for (int ui = 0; ui < 3; ++ui) {
        int u  = wave + 8 * ui;            // scalar: 0..23
        int h  = u >> 2;
        int mt = u & 3;
        const float* rpbh = rpbf + h * (NTOK * NTOK);
        const __bf16* qh = qs + h * 64 * QSTR;
        const __bf16* kh = ks + h * 64 * QSTR;
        const __bf16* vh = vt + h * 32 * VSTR;

        // scores S = (q*scale) @ k^T for one 16-row block (K = 32 = head dim)
        v16bf Aq = load_fragA(qh, QSTR, mt * 16, lane, 0);
        v8f S[4];
        #pragma unroll
        for (int nt = 0; nt < 4; ++nt) {
            v16bf Bk = load_fragB(kh, QSTR, nt * 16, lane, 0);
            S[nt] = __builtin_amdgcn_wmma_f32_16x16x32_bf16(false, Aq, false, Bk,
                                                            (short)0, vz8(), false, false);
        }
        // + relative position bias + window mask
        #pragma unroll
        for (int nt = 0; nt < 4; ++nt)
            #pragma unroll
            for (int r = 0; r < 8; ++r) {
                int m = mt * 16 + r + rhi;
                int n = nt * 16 + n_lo;
                S[nt][r] += rpbh[m * 64 + n] + maskw[m * 64 + n];
            }
        // softmax over n (row m lives in one 16-lane half -> xor 8/4/2/1 reduces)
        float rmax[8], rsum[8];
        #pragma unroll
        for (int r = 0; r < 8; ++r)
            rmax[r] = fmaxf(fmaxf(S[0][r], S[1][r]), fmaxf(S[2][r], S[3][r]));
        #pragma unroll
        for (int off = 8; off >= 1; off >>= 1)
            #pragma unroll
            for (int r = 0; r < 8; ++r)
                rmax[r] = fmaxf(rmax[r], __shfl_xor(rmax[r], off, 32));
        #pragma unroll
        for (int r = 0; r < 8; ++r) rsum[r] = 0.f;
        #pragma unroll
        for (int nt = 0; nt < 4; ++nt)
            #pragma unroll
            for (int r = 0; r < 8; ++r) {
                float p = __expf(S[nt][r] - rmax[r]);
                S[nt][r] = p;
                rsum[r] += p;
            }
        #pragma unroll
        for (int off = 8; off >= 1; off >>= 1)
            #pragma unroll
            for (int r = 0; r < 8; ++r)
                rsum[r] += __shfl_xor(rsum[r], off, 32);
        #pragma unroll
        for (int r = 0; r < 8; ++r) {
            float inv = 1.0f / rsum[r];
            int ml = r + rhi;              // local row 0..15 in per-wave scratch
            #pragma unroll
            for (int nt = 0; nt < 4; ++nt)
                ph[ml * PSTR + nt * 16 + n_lo] = (__bf16)(S[nt][r] * inv);
        }

        // O(rows mt) = P(rows mt) @ V  (K = 64 tokens -> two WMMA k-steps)
        v16bf A0 = load_fragA(ph, PSTR, 0, lane, 0);
        v16bf A1 = load_fragA(ph, PSTR, 0, lane, 32);
        #pragma unroll
        for (int nd = 0; nd < 2; ++nd) {
            v16bf B0 = load_fragB(vh, VSTR, nd * 16, lane, 0);
            v16bf B1 = load_fragB(vh, VSTR, nd * 16, lane, 32);
            v8f o = __builtin_amdgcn_wmma_f32_16x16x32_bf16(false, A0, false, B0,
                                                            (short)0, vz8(), false, false);
            o = __builtin_amdgcn_wmma_f32_16x16x32_bf16(false, A1, false, B1,
                                                        (short)0, o, false, false);
            __bf16* p = xs + (mt * 16 + rhi) * XSTR + h * HD + nd * 16 + n_lo;
            #pragma unroll
            for (int r = 0; r < 8; ++r)
                p[r * XSTR] = (__bf16)o[r];    // O staging reuses xs (x is dead)
        }
    }
    __syncthreads();

    // -------- Phase E: Y = O @ Wproj + b -> global f32 --------
    float* outw = out + (size_t)w * NTOK * DIM;
    #pragma unroll 1
    for (int ti = 0; ti < 6; ++ti) {
        int t  = wave + 8 * ti;            // scalar: 48 tiles = 4 x 12
        int mt = t / 12, nt = t % 12;
        v8f acc = vz8();
        #pragma unroll
        for (int kt = 0; kt < 6; ++kt) {
            v16bf A = load_fragA(xs, XSTR, mt * 16, lane, kt * 32);
            v16bf B = load_wblk(wproj, kt * 12 + nt, lane);
            acc = __builtin_amdgcn_wmma_f32_16x16x32_bf16(false, A, false, B,
                                                          (short)0, acc, false, false);
        }
        int col = nt * 16 + n_lo;
        float bias = bproj[col];
        float* p = outw + (mt * 16 + rhi) * DIM + col;
        #pragma unroll
        for (int r = 0; r < 8; ++r)
            p[r * DIM] = acc[r] + bias;
    }
}

extern "C" void kernel_launch(void* const* d_in, const int* in_sizes, int n_in,
                              void* d_out, int out_size, void* d_ws, size_t ws_size,
                              hipStream_t stream) {
    const float* x      = (const float*)d_in[0];
    const float* mask   = (const float*)d_in[1];
    const float* qkv_w  = (const float*)d_in[2];
    const float* qkv_b  = (const float*)d_in[3];
    const float* proj_w = (const float*)d_in[4];
    const float* proj_b = (const float*)d_in[5];
    const float* rpb_t  = (const float*)d_in[6];
    const int*   rel_i  = (const int*)d_in[7];

    char* ws = (char*)d_ws;
    __bf16* wqkv_p  = (__bf16*)ws;               // 216 blocks * 1024 B = 221184
    __bf16* wproj_p = (__bf16*)(ws + 221184);    //  72 blocks * 1024 B =  73728
    float*  rpbf    = (float*)(ws + 294912);     //  6*64*64*4          =  98304

    pack_w_kernel<<<216, 32, 0, stream>>>(qkv_w, wqkv_p, QKV_OUT, 36);
    pack_w_kernel<<<72, 32, 0, stream>>>(proj_w, wproj_p, DIM, 12);
    rpb_kernel<<<96, 256, 0, stream>>>(rpb_t, rel_i, rpbf);

    win_attn_kernel<<<NWIN, 256, SMEM_BYTES, stream>>>(
        x, mask, wqkv_p, qkv_b, wproj_p, proj_b, rpbf, (float*)d_out);
}
